// SelfAttention2D_1597727834424
// MI455X (gfx1250) — matrix operand order repaired and checked by reading, not compiled, not run on hardware
//
#include <hip/hip_runtime.h>
#include <hip/hip_bf16.h>

#define HW   4096
#define CCH  256
#define DD   32
#define NB   4

typedef __attribute__((ext_vector_type(16))) _Float16 v16h;
typedef __attribute__((ext_vector_type(8)))  _Float16 v8h;
typedef __attribute__((ext_vector_type(8)))  float    v8f;
typedef __attribute__((ext_vector_type(4)))  unsigned int v4u;
typedef __attribute__((ext_vector_type(8)))  int      v8i;
typedef __attribute__((ext_vector_type(4)))  int      v4i;

#if __has_builtin(__builtin_amdgcn_tensor_load_to_lds) && __has_builtin(__builtin_amdgcn_s_wait_tensorcnt)
#define ATTN_TDM 1
#else
#define ATTN_TDM 0
#endif

__device__ __forceinline__ v8f wmma_f16(v16h a, v16h b, v8f c) {
    // (neg_a, A, neg_b, B, c_mod, C, reuse_a, reuse_b)
    return __builtin_amdgcn_wmma_f32_16x16x32_f16(false, a, false, b, (short)0, c, false, false);
}

// Reduce across the 16 lanes of each half-wave (wave32; masks <16 stay in-half).
__device__ __forceinline__ float rowmax16(float v) {
    for (int m = 1; m < 16; m <<= 1) v = fmaxf(v, __shfl_xor(v, m, 32));
    return v;
}
__device__ __forceinline__ float rowsum16(float v) {
    for (int m = 1; m < 16; m <<= 1) v += __shfl_xor(v, m, 32);
    return v;
}

// A-fragment (16x32 f16) K position for VGPR-pair vg, low half of pair.
__device__ __forceinline__ int aK(int vg, int hi) {
    return (hi << 4) + (vg << 1);
}

#if ATTN_TDM
// Generic pointer to a __shared__ object: flat addr [31:0] is the LDS offset
// (LDS aperture truncates to low 32 bits per CDNA5 aperture rules).
__device__ __forceinline__ unsigned lds_offset(const void* p) {
    return (unsigned)(size_t)p;
}

// TDM: DMA a 2D tile [256 rows x 32 halfs] out of the [256 x 4096]-half V
// tensor into LDS, padding each 64B row with 16B -> 80B LDS row stride.
// D# fields per CDNA5 ISA ch.8 (group0/group1 bit layout); groups 2/3 zero
// (2D tensor, tile_dim2/3 = 0). clang-23 builtin: 6 args with an extra
// int32x8 group before the cache-policy immediate.
__device__ __forceinline__ void tdm_load_vtile(const _Float16* gsrc, unsigned lds_off) {
    unsigned long long ga = (unsigned long long)(size_t)gsrc;
    v4u g0 = { 1u,                                      // count=1, user mode
               lds_off,                                 // lds_addr [63:32]
               (unsigned)ga,                            // global_addr [95:64]
               (unsigned)((ga >> 32) & 0x01FFFFFFu)     // global_addr [120:96]
                   | (2u << 30) };                      // type=2 ("image") [127:126]
    v8i g1 = { (int)((1u << 16)    // data_size = 1 (2 bytes)
                   | (1u << 20)    // pad_enable
                   | (3u << 22)    // pad_interval: 16 DWORDs (64B) between pads
                   | (3u << 25)),  // pad_amount: 4 DWORDs (16B)
               (int)(4096u << 16), // tensor_dim0[15:0]=4096 at bits 63:48
               (int)(256u << 16),  // tensor_dim1[15:0]=256 at bits 95:80
               (int)(32u << 16),   // tile_dim0=32 at bits 127:112
               256,                // tile_dim1=256 at bits 143:128
               4096,               // tensor_dim0_stride[31:0]=4096 at bits 191:160
               0, 0 };
    v4i g2 = { 0, 0, 0, 0 };
    v4i g3 = { 0, 0, 0, 0 };
    v8i g4 = { 0, 0, 0, 0, 0, 0, 0, 0 };
    __builtin_amdgcn_tensor_load_to_lds(g0, g1, g2, g3, g4, 0);
}
#endif

// ---------------------------------------------------------------------------
// Kernel 0: pack weights to f16, concat biases.
__global__ void __launch_bounds__(256) prep_kernel(
    const float* __restrict__ Wq, const float* __restrict__ Wk,
    const float* __restrict__ Wv, const float* __restrict__ Wo,
    const float* __restrict__ bq, const float* __restrict__ bk,
    const float* __restrict__ bv,
    _Float16* __restrict__ Wqkv_h, _Float16* __restrict__ Wo_h,
    float* __restrict__ bias_cat)
{
    int i = blockIdx.x * 256 + threadIdx.x;
    if (i < 320 * 256) {
        float v;
        if (i < 32 * 256)       v = Wq[i];
        else if (i < 64 * 256)  v = Wk[i - 32 * 256];
        else                    v = Wv[i - 64 * 256];
        Wqkv_h[i] = (_Float16)v;
    }
    if (i < 256 * 256) Wo_h[i] = (_Float16)Wo[i];
    if (i < 320) bias_cat[i] = (i < 32) ? bq[i] : (i < 64 ? bk[i - 32] : bv[i - 64]);
}

// ---------------------------------------------------------------------------
// Kernel 1: fused QKV projection. Block = 256 thr (8 waves), 32 pixels/block.
// Wave w: pixel half (w&1)*16, output-channel set (w>>1)*80 .. +79 (5 groups).
__global__ void __launch_bounds__(256) qkv_kernel(
    const float* __restrict__ x, const _Float16* __restrict__ Wqkv_h,
    const float* __restrict__ bias_cat,
    _Float16* __restrict__ qbuf, _Float16* __restrict__ kbuf,
    _Float16* __restrict__ vbuf)
{
    __shared__ _Float16 xt[32 * 264];   // [px][ch], 264-half stride (132 words)
    const int tid = threadIdx.x;
    const size_t gpx = (size_t)blockIdx.x * 32;
    const int b = (int)(gpx / HW);
    const int px0 = (int)(gpx % HW);

    // Stage X tile transposed into LDS as f16: thread t owns channel t.
    {
        const float* src = x + ((size_t)b * CCH + tid) * HW + px0;
        #pragma unroll 4
        for (int p = 0; p < 32; ++p) xt[p * 264 + tid] = (_Float16)src[p];
    }
    __syncthreads();

    const int w = tid >> 5, lane = tid & 31, r = lane & 15, hi = lane >> 4;
    const int pxh = (w & 1) * 16;
    const int colset = w >> 1;          // 0..3

    v8f acc[5];
    #pragma unroll
    for (int g = 0; g < 5; ++g) { v8f z = {}; acc[g] = z; }

    for (int k0 = 0; k0 < 256; k0 += 32) {
        union { v16h h; unsigned u[8]; } A;
        const _Float16* arow = &xt[(pxh + r) * 264 + k0];
        #pragma unroll
        for (int vg = 0; vg < 8; ++vg)
            A.u[vg] = *(const unsigned*)(arow + aK(vg, hi));
        #pragma unroll
        for (int g = 0; g < 5; ++g) {
            const int oc0 = colset * 80 + g * 16;
            v16h Bf = *(const v16h*)(Wqkv_h + (size_t)(oc0 + r) * 256 + k0 + hi * 16);
            acc[g] = wmma_f16(A.h, Bf, acc[g]);
        }
    }

    #pragma unroll
    for (int g = 0; g < 5; ++g) {
        const int oc = colset * 80 + g * 16 + r;
        const float bc = bias_cat[oc];
        #pragma unroll
        for (int vg = 0; vg < 8; ++vg) {
            const int m = vg + hi * 8;
            const size_t px = (size_t)px0 + pxh + m;
            const _Float16 h = (_Float16)(acc[g][vg] + bc);
            if (oc < 32)      qbuf[((size_t)b * HW + px) * DD + oc] = h;
            else if (oc < 64) kbuf[((size_t)b * HW + px) * DD + (oc - 32)] = h;
            else              vbuf[((size_t)b * CCH + (oc - 64)) * HW + px] = h;
        }
    }
}

// ---------------------------------------------------------------------------
// Kernel 2: flash attention. Block = 128 thr (4 waves) -> 64 query rows.
// d=32 == WMMA K: one v_wmma per 16-key score tile. V tile double-buffered in
// LDS via the Tensor Data Mover (DMA of tile i+1 overlaps compute of tile i).
__global__ void __launch_bounds__(128) attn_kernel(
    const _Float16* __restrict__ qbuf, const _Float16* __restrict__ kbuf,
    const _Float16* __restrict__ vbuf, const float* __restrict__ mask,
    _Float16* __restrict__ attnout)
{
    __shared__ _Float16 vtile[2][CCH * 40];  // [ch][j], 40-half (80 B) rows
    __shared__ _Float16 ptile[4][16 * 34];   // per-wave P transpose, 34-half rows

    const int tid = threadIdx.x, w = tid >> 5, lane = tid & 31;
    const int r = lane & 15, hi = lane >> 4;
    const size_t gq = (size_t)blockIdx.x * 64;
    const int b = (int)(gq / HW);
    const int q0 = (int)(gq % HW) + w * 16;
    const _Float16* vbase = vbuf + (size_t)b * CCH * HW;

    // Q A-fragment: rows are contiguous 32-half vectors in qbuf.
    union AU { v16h h; unsigned u[8]; };
    AU Qa;
    {
        const _Float16* qrow = qbuf + ((size_t)b * HW + q0 + r) * DD;
        #pragma unroll
        for (int vg = 0; vg < 8; ++vg)
            Qa.u[vg] = *(const unsigned*)(qrow + aK(vg, hi));
    }

    float mi[8], mrow[8], lrow[8];
    #pragma unroll
    for (int vg = 0; vg < 8; ++vg) {
        mi[vg]   = mask[(size_t)b * HW + q0 + vg + hi * 8];
        mrow[vg] = -3.0e38f;
        lrow[vg] = 0.0f;
    }
    v8f O[16];
    #pragma unroll
    for (int cg = 0; cg < 16; ++cg) { v8f z = {}; O[cg] = z; }

#if ATTN_TDM
    if (w == 0)  // one DMA per workgroup; TENSORcnt is per-wave
        tdm_load_vtile(vbase, lds_offset(&vtile[0][0]));
#endif

    for (int j0 = 0; j0 < HW; j0 += 32) {
        const int buf = (j0 >> 5) & 1;
#if ATTN_TDM
        __builtin_amdgcn_s_wait_tensorcnt(0);  // no-op for waves 1..3
        __syncthreads();                       // tile j0 visible; tile buf^1 free
        if (w == 0 && (j0 + 32 < HW))
            tdm_load_vtile(vbase + j0 + 32, lds_offset(&vtile[buf ^ 1][0]));
#else
        __syncthreads();
        // Stage V tile [256 ch x 32 j] f16 -> LDS (4 threads per row, 16 B each)
        #pragma unroll
        for (int rr0 = 0; rr0 < 256; rr0 += 32) {
            const int row = rr0 + (tid >> 2), part = tid & 3;
            const v8h vv = *(const v8h*)(vbase + (size_t)row * HW + j0 + part * 8);
            *(v8h*)&vtile[buf][row * 40 + part * 8] = vv;
        }
        if (j0 + 32 < HW) {
            #pragma unroll
            for (int rr0 = 0; rr0 < 256; rr0 += 128)
                __builtin_prefetch(vbase + (size_t)(rr0 + tid) * HW + j0 + 32, 0, 0);
        }
        __syncthreads();
#endif

        // S = Q x K^T for 32 keys: two WMMAs (K rows contiguous in kbuf).
        const v16h Kb1 = *(const v16h*)(kbuf + ((size_t)b * HW + j0 + r) * DD + hi * 16);
        const v16h Kb2 = *(const v16h*)(kbuf + ((size_t)b * HW + j0 + 16 + r) * DD + hi * 16);
        v8f S1 = {}, S2 = {};
        S1 = wmma_f16(Qa.h, Kb1, S1);
        S2 = wmma_f16(Qa.h, Kb2, S2);

        const float mj1 = mask[(size_t)b * HW + j0 + r];
        const float mj2 = mask[(size_t)b * HW + j0 + 16 + r];

        float corr[8];
        #pragma unroll
        for (int vg = 0; vg < 8; ++vg) {
            const float mm1 = mi[vg] * mj1, mm2 = mi[vg] * mj2;
            const float s1 = (mm1 == 0.0f) ? -3.0e38f : S1[vg] * mm1;
            const float s2 = (mm2 == 0.0f) ? -3.0e38f : S2[vg] * mm2;
            const float rmax = rowmax16(fmaxf(s1, s2));
            const float mnew = fmaxf(mrow[vg], rmax);
            const float c  = (mrow[vg] > -1.0e30f) ? __expf(mrow[vg] - mnew) : 0.0f;
            const float p1 = (s1 > -1.0e30f) ? __expf(s1 - mnew) : 0.0f;
            const float p2 = (s2 > -1.0e30f) ? __expf(s2 - mnew) : 0.0f;
            lrow[vg] = lrow[vg] * c + rowsum16(p1 + p2);
            mrow[vg] = mnew;
            corr[vg] = c;
            const int m = vg + hi * 8;         // D-layout row
            ptile[w][m * 34 + r]      = (_Float16)p1;
            ptile[w][m * 34 + 16 + r] = (_Float16)p2;
        }
        #pragma unroll
        for (int cg = 0; cg < 16; ++cg)
            #pragma unroll
            for (int vg = 0; vg < 8; ++vg) O[cg][vg] *= corr[vg];

        // Re-load P in A-layout (wave-local LDS transpose).
        AU Pa;
        {
            const _Float16* prow = &ptile[w][r * 34];
            #pragma unroll
            for (int vg = 0; vg < 8; ++vg)
                Pa.u[vg] = *(const unsigned*)(prow + aK(vg, hi));
        }

        // O += P x V^T : 16 channel-group WMMAs, V fragments from LDS.
        #pragma unroll
        for (int cg = 0; cg < 16; ++cg) {
            union { v16h v; v8h h2[2]; } Vb;
            const _Float16* vrow = &vtile[buf][(cg * 16 + r) * 40 + hi * 16];
            Vb.h2[0] = *(const v8h*)(vrow);
            Vb.h2[1] = *(const v8h*)(vrow + 8);
            O[cg] = wmma_f16(Pa.h, Vb.v, O[cg]);
        }
    }

    #pragma unroll
    for (int vg = 0; vg < 8; ++vg)
        lrow[vg] = (lrow[vg] > 0.0f) ? 1.0f / lrow[vg] : 0.0f;
    #pragma unroll
    for (int cg = 0; cg < 16; ++cg) {
        const int c = cg * 16 + r;
        #pragma unroll
        for (int vg = 0; vg < 8; ++vg) {
            const int m = vg + hi * 8;
            attnout[((size_t)b * HW + q0 + m) * CCH + c] = (_Float16)(O[cg][vg] * lrow[vg]);
        }
    }
}

// ---------------------------------------------------------------------------
// Kernel 3: out-projection (WMMA) + gamma*out + x residual + channel LayerNorm.
// Block = 128 thr (4 waves); wave handles 16 pixels x all 256 channels.
__global__ void __launch_bounds__(128) oproj_kernel(
    const _Float16* __restrict__ attnout, const _Float16* __restrict__ Wo_h,
    const float* __restrict__ bo, const float* __restrict__ gamma,
    const float* __restrict__ x, const float* __restrict__ ln_w,
    const float* __restrict__ ln_b, float* __restrict__ out)
{
    const int tid = threadIdx.x, w = tid >> 5, lane = tid & 31;
    const int r = lane & 15, hi = lane >> 4;
    const size_t gpx = (size_t)blockIdx.x * 64;
    const int b = (int)(gpx / HW);
    const int px0 = (int)(gpx % HW) + w * 16;

    v8f acc[16];
    #pragma unroll
    for (int cg = 0; cg < 16; ++cg) { v8f z = {}; acc[cg] = z; }

    for (int k0 = 0; k0 < 256; k0 += 32) {
        union { v16h h; unsigned u[8]; } A;
        const _Float16* arow = attnout + ((size_t)b * HW + px0 + r) * CCH + k0;
        #pragma unroll
        for (int vg = 0; vg < 8; ++vg)
            A.u[vg] = *(const unsigned*)(arow + aK(vg, hi));
        #pragma unroll
        for (int cg = 0; cg < 16; ++cg) {
            v16h Bf = *(const v16h*)(Wo_h + (size_t)(cg * 16 + r) * 256 + k0 + hi * 16);
            acc[cg] = wmma_f16(A.h, Bf, acc[cg]);
        }
    }

    const float g = gamma[0];
    #pragma unroll
    for (int cg = 0; cg < 16; ++cg) {
        const int c = cg * 16 + r;
        const float bc = bo[c];
        #pragma unroll
        for (int vg = 0; vg < 8; ++vg) {
            const int m = vg + hi * 8;
            const int px = px0 + m;
            acc[cg][vg] = g * (acc[cg][vg] + bc) + x[((size_t)b * CCH + c) * HW + px];
        }
    }
    // LayerNorm over channels per pixel row (row = vg + 8*hi).
    float mu[8], rstd[8];
    #pragma unroll
    for (int vg = 0; vg < 8; ++vg) {
        float s = 0.0f;
        #pragma unroll
        for (int cg = 0; cg < 16; ++cg) s += acc[cg][vg];
        mu[vg] = rowsum16(s) * (1.0f / 256.0f);
    }
    #pragma unroll
    for (int vg = 0; vg < 8; ++vg) {
        float s = 0.0f;
        #pragma unroll
        for (int cg = 0; cg < 16; ++cg) { const float d = acc[cg][vg] - mu[vg]; s += d * d; }
        rstd[vg] = rsqrtf(rowsum16(s) * (1.0f / 256.0f) + 1e-5f);
    }
    #pragma unroll
    for (int cg = 0; cg < 16; ++cg) {
        const int c = cg * 16 + r;
        const float lw = ln_w[c], lb = ln_b[c];
        #pragma unroll
        for (int vg = 0; vg < 8; ++vg) {
            const int m = vg + hi * 8;
            const int px = px0 + m;
            out[((size_t)b * CCH + c) * HW + px] = (acc[cg][vg] - mu[vg]) * rstd[vg] * lw + lb;
        }
    }
}

// ---------------------------------------------------------------------------
extern "C" void kernel_launch(void* const* d_in, const int* in_sizes, int n_in,
                              void* d_out, int out_size, void* d_ws, size_t ws_size,
                              hipStream_t stream) {
    const float* x     = (const float*)d_in[0];
    const float* mask  = (const float*)d_in[1];
    const float* Wq    = (const float*)d_in[2];
    const float* bq    = (const float*)d_in[3];
    const float* Wk    = (const float*)d_in[4];
    const float* bk    = (const float*)d_in[5];
    const float* Wv    = (const float*)d_in[6];
    const float* bv    = (const float*)d_in[7];
    const float* Wo    = (const float*)d_in[8];
    const float* bo    = (const float*)d_in[9];
    const float* gamma = (const float*)d_in[10];
    const float* ln_w  = (const float*)d_in[11];
    const float* ln_b  = (const float*)d_in[12];
    float* out = (float*)d_out;

    char* ws = (char*)d_ws;
    size_t off = 0;
    auto take = [&](size_t bytes) { char* p = ws + off; off += (bytes + 255) & ~(size_t)255; return p; };
    _Float16* Wqkv_h   = (_Float16*)take((size_t)320 * 256 * 2);
    _Float16* Wo_h     = (_Float16*)take((size_t)256 * 256 * 2);
    float*    bias_cat = (float*)   take((size_t)320 * 4);
    _Float16* qb       = (_Float16*)take((size_t)NB * HW * DD * 2);
    _Float16* kb       = (_Float16*)take((size_t)NB * HW * DD * 2);
    _Float16* vb       = (_Float16*)take((size_t)NB * CCH * HW * 2);
    _Float16* ao       = (_Float16*)take((size_t)NB * HW * CCH * 2);

    prep_kernel<<<320, 256, 0, stream>>>(Wq, Wk, Wv, Wo, bq, bk, bv,
                                         Wqkv_h, Wo_h, bias_cat);
    qkv_kernel<<<NB * HW / 32, 256, 0, stream>>>(x, Wqkv_h, bias_cat, qb, kb, vb);
    attn_kernel<<<NB * HW / 64, 128, 0, stream>>>(qb, kb, vb, mask, ao);
    oproj_kernel<<<NB * HW / 64, 128, 0, stream>>>(ao, Wo_h, bo, gamma, x, ln_w, ln_b, out);
}
